// GraphAttentionNetwork_2697239461975
// MI455X (gfx1250) — matrix operand order repaired and checked by reading, not compiled, not run on hardware
//
#include <hip/hip_runtime.h>
#include <hip/hip_bf16.h>

// ---------------------------------------------------------------------------
// GAT forward on MI455X (gfx1250, wave32). GEMMs via v_wmma_f32_16x16x32_f16.
// ---------------------------------------------------------------------------

typedef __attribute__((ext_vector_type(16))) _Float16 v16h;
typedef __attribute__((ext_vector_type(8)))  float    v8f;

#define GN      50000
#define GE      400000
#define GD      128
#define GH      2
#define GHOPS   3
#define GC      40

#define AS_LDA  36   // 64x32 f16 A tile, padded row stride (halves); 72B rows -> 8B aligned
#define BS_LDA  36   // 32x32 f16 B^T tile, padded row stride (halves)

// ----------------------------- helpers -------------------------------------

__device__ __forceinline__ float waveSum(float v) {
    #pragma unroll
    for (int off = 16; off > 0; off >>= 1) v += __shfl_xor(v, off, 32);
    return v;
}

// float atomic max via signed/unsigned integer monotonicity trick.
__device__ __forceinline__ void atomicMaxF(float* addr, float val) {
    if (val >= 0.0f) atomicMax((int*)addr, __float_as_int(val));
    else             atomicMin((unsigned int*)addr, (unsigned int)__float_as_int(val));
}

// ----------------------------- WMMA GEMM -----------------------------------
// C[M x Nc] = act(A[M x K] @ W[K x Nc] + bias), row-major. K = KSTEPS*32.
// Block: 128 threads = 4 waves, macro-tile 64 rows x 32 cols (2 WMMA / wave /
// K-step). All tile loads are UNCONDITIONAL (clamped addresses); only final
// stores are guarded -> EXEC stays fully converged around the WMMAs.
template <int KSTEPS>
__global__ __launch_bounds__(128)
void gemm_bias_act_wmma(const float* __restrict__ A, const float* __restrict__ W,
                        const float* __restrict__ bias, float* __restrict__ C,
                        int M, int Nc, int doRelu) {
    constexpr int K = KSTEPS * 32;
    __shared__ _Float16 As[64 * AS_LDA];
    __shared__ _Float16 Bs[32 * BS_LDA];   // stored transposed: Bs[n][k]

    const int tid  = threadIdx.x;
    const int lane = tid & 31;
    const int wave = tid >> 5;     // 0..3 -> row strip
    const int g    = lane >> 4;    // lane group (0/1)
    const int ln   = lane & 15;

    const int rowBase = blockIdx.x * 64;
    const int colBase = blockIdx.y * 32;

    // ---- Precompute per-thread staging pointers (advanced per K-step) ----
    const float* aPtr[4];
    int aLds[4];
    #pragma unroll
    for (int i = 0; i < 4; ++i) {
        int idx4 = tid + i * 128;          // 0..511 quads of the 64x32 tile
        int r    = idx4 >> 3;              // 8 quads per 32-wide row
        int c4   = (idx4 & 7) * 4;
        int gr   = rowBase + r;
        if (gr >= M) gr = M - 1;           // clamp: garbage rows are store-guarded
        aPtr[i] = A + (long)gr * K + c4;
        aLds[i] = r * AS_LDA + c4;
    }
    const float* bPtr[4];
    int bLds[4];
    #pragma unroll
    for (int i = 0; i < 4; ++i) {
        int idx2 = tid + i * 128;          // 0..511 (k-pair, col) of 32x32 tile
        int nn   = idx2 & 31;
        int kp   = idx2 >> 5;              // 0..15 -> k = 2*kp
        int gc   = colBase + nn;
        if (gc >= Nc) gc = Nc - 1;         // clamp: garbage cols are store-guarded
        bPtr[i] = W + (long)(2 * kp) * Nc + gc;
        bLds[i] = nn * BS_LDA + 2 * kp;
    }
    const long bStep = 32L * Nc;

    v8f acc0 = {};
    v8f acc1 = {};

    #pragma unroll
    for (int ks = 0; ks < KSTEPS; ++ks) {
        // ---- Phase 1: issue ALL global loads back-to-back ----
        float4 va[4];
        #pragma unroll
        for (int i = 0; i < 4; ++i) { va[i] = *(const float4*)aPtr[i]; aPtr[i] += 32; }
        float vb0[4], vb1[4];
        #pragma unroll
        for (int i = 0; i < 4; ++i) {
            vb0[i] = bPtr[i][0];
            vb1[i] = bPtr[i][Nc];
            bPtr[i] += bStep;
        }
        // ---- Phase 2: convert to f16 and store to LDS ----
        #pragma unroll
        for (int i = 0; i < 4; ++i) {
            union { _Float16 h[4]; uint2 u2; } p;
            p.h[0] = (_Float16)va[i].x; p.h[1] = (_Float16)va[i].y;
            p.h[2] = (_Float16)va[i].z; p.h[3] = (_Float16)va[i].w;
            *(uint2*)(&As[aLds[i]]) = p.u2;
        }
        #pragma unroll
        for (int i = 0; i < 4; ++i) {
            union { _Float16 h[2]; unsigned u; } p;
            p.h[0] = (_Float16)vb0[i]; p.h[1] = (_Float16)vb1[i];
            *(unsigned*)(&Bs[bLds[i]]) = p.u;
        }
        __syncthreads();

        // ---- Fragments per documented CDNA5 16-bit layouts ----
        union { v16h v; unsigned u[8]; } af, bf0, bf1;
        const int aRow = wave * 16 + ln;
        #pragma unroll
        for (int j = 0; j < 8; ++j) {
            int ka = (j < 4) ? (g * 8 + 2 * j) : (16 + g * 8 + 2 * (j - 4));
            af.u[j] = *(const unsigned*)(&As[aRow * AS_LDA + ka]);
            int kb  = ka;                  // B: same K order as A (K=8g..8g+7, 16+8g..16+8g+7)
            bf0.u[j] = *(const unsigned*)(&Bs[ln * BS_LDA + kb]);
            bf1.u[j] = *(const unsigned*)(&Bs[(16 + ln) * BS_LDA + kb]);
        }
        acc0 = __builtin_amdgcn_wmma_f32_16x16x32_f16(
                   false, af.v, false, bf0.v, (short)0, acc0, false, false);
        acc1 = __builtin_amdgcn_wmma_f32_16x16x32_f16(
                   false, af.v, false, bf1.v, (short)0, acc1, false, false);
        __syncthreads();
    }

    // C/D layout: VGPR j, lane group g -> row m = j + 8*g, col = lane&15.
    const int col0 = colBase + ln;
    const int col1 = colBase + 16 + ln;
    const float bv0 = bias[(col0 < Nc) ? col0 : (Nc - 1)];
    const float bv1 = bias[(col1 < Nc) ? col1 : (Nc - 1)];
    #pragma unroll
    for (int j = 0; j < 8; ++j) {
        int row = rowBase + wave * 16 + (j + 8 * g);
        if (row < M) {
            if (col0 < Nc) {
                float v = acc0[j] + bv0;
                if (doRelu) v = fmaxf(v, 0.0f);
                C[(long)row * Nc + col0] = v;
            }
            if (col1 < Nc) {
                float v = acc1[j] + bv1;
                if (doRelu) v = fmaxf(v, 0.0f);
                C[(long)row * Nc + col1] = v;
            }
        }
    }
}

// ------------------------- per-hop init ------------------------------------
__global__ void hop_init(float* __restrict__ smax, float* __restrict__ ssum,
                         float* __restrict__ agg, int n2, int nAgg) {
    int i = blockIdx.x * blockDim.x + threadIdx.x;
    if (i < nAgg) agg[i] = 0.0f;
    if (i < n2) { smax[i] = -__builtin_huge_valf(); ssum[i] = 0.0f; }
}

// ---------------- per-node attention projections ---------------------------
// aS[n,h] = x[n,:] . attW[h, 0:128] ; aR[n,h] = x[n,:] . attW[h, 128:256]
__global__ __launch_bounds__(256)
void alpha_kernel(const float* __restrict__ x, const float* __restrict__ wAtt,
                  float* __restrict__ aS, float* __restrict__ aR, int N) {
    const int lane = threadIdx.x & 31;
    const int node = blockIdx.x * 8 + (threadIdx.x >> 5);
    if (node >= N) return;
    const int d = lane * 4;
    const float4 xv = *(const float4*)(x + (long)node * GD + d);
    const float4 w0 = *(const float4*)(wAtt + d);
    const float4 w1 = *(const float4*)(wAtt + GD + d);
    const float4 w2 = *(const float4*)(wAtt + 2 * GD + d);
    const float4 w3 = *(const float4*)(wAtt + 3 * GD + d);
    float s0 = xv.x * w0.x + xv.y * w0.y + xv.z * w0.z + xv.w * w0.w;
    float r0 = xv.x * w1.x + xv.y * w1.y + xv.z * w1.z + xv.w * w1.w;
    float s1 = xv.x * w2.x + xv.y * w2.y + xv.z * w2.z + xv.w * w2.w;
    float r1 = xv.x * w3.x + xv.y * w3.y + xv.z * w3.z + xv.w * w3.w;
    s0 = waveSum(s0); r0 = waveSum(r0); s1 = waveSum(s1); r1 = waveSum(r1);
    if (lane == 0) {
        aS[node * 2 + 0] = s0; aS[node * 2 + 1] = s1;
        aR[node * 2 + 0] = r0; aR[node * 2 + 1] = r1;
    }
}

// --------------- edge pass 1: leaky-relu logits + segment max --------------
__global__ void edge_logits_max(const float* __restrict__ aS, const float* __restrict__ aR,
                                const int* __restrict__ sIdx, const int* __restrict__ rIdx,
                                const float* __restrict__ attB,
                                float* __restrict__ nl, float* __restrict__ smax, int E) {
    int e = blockIdx.x * blockDim.x + threadIdx.x;
    if (e >= E) return;
    int s = sIdx[e], r = rIdx[e];
    float l0 = aS[s * 2 + 0] + aR[r * 2 + 0] + attB[0];
    float l1 = aS[s * 2 + 1] + aR[r * 2 + 1] + attB[1];
    l0 = (l0 > 0.0f) ? l0 : 0.2f * l0;
    l1 = (l1 > 0.0f) ? l1 : 0.2f * l1;
    nl[e * 2 + 0] = l0; nl[e * 2 + 1] = l1;
    atomicMaxF(&smax[r * 2 + 0], l0);
    atomicMaxF(&smax[r * 2 + 1], l1);
}

// --------------- edge pass 2: exp + segment sum ----------------------------
__global__ void edge_exp_sum(float* __restrict__ wbuf, const int* __restrict__ rIdx,
                             const float* __restrict__ smax, float* __restrict__ ssum, int E) {
    int e = blockIdx.x * blockDim.x + threadIdx.x;
    if (e >= E) return;
    int r = rIdx[e];
    float w0 = __expf(wbuf[e * 2 + 0] - smax[r * 2 + 0]);
    float w1 = __expf(wbuf[e * 2 + 1] - smax[r * 2 + 1]);
    wbuf[e * 2 + 0] = w0; wbuf[e * 2 + 1] = w1;
    atomicAdd(&ssum[r * 2 + 0], w0);
    atomicAdd(&ssum[r * 2 + 1], w1);
}

// --------------- edge pass 3: normalized weighted scatter-add --------------
// agg[r, h*128 + d] += (w[e,h]/ssum[r,h]) * x[s, d]   (head-major concat)
__global__ __launch_bounds__(256)
void edge_aggregate(const float* __restrict__ x,
                    const int* __restrict__ sIdx, const int* __restrict__ rIdx,
                    const float* __restrict__ wbuf, const float* __restrict__ ssum,
                    float* __restrict__ agg, int E) {
    const int lane = threadIdx.x & 31;
    const int e = blockIdx.x * 8 + (threadIdx.x >> 5);
    if (e >= E) return;
    int s = sIdx[e], r = rIdx[e];
    float c0 = wbuf[e * 2 + 0] / ssum[r * 2 + 0];
    float c1 = wbuf[e * 2 + 1] / ssum[r * 2 + 1];
    const int d = lane * 4;
    const float4 xv = *(const float4*)(x + (long)s * GD + d);
    float* a0 = agg + (long)r * (2 * GD) + d;
    atomicAdd(a0 + 0, c0 * xv.x);
    atomicAdd(a0 + 1, c0 * xv.y);
    atomicAdd(a0 + 2, c0 * xv.z);
    atomicAdd(a0 + 3, c0 * xv.w);
    float* a1 = a0 + GD;
    atomicAdd(a1 + 0, c1 * xv.x);
    atomicAdd(a1 + 1, c1 * xv.y);
    atomicAdd(a1 + 2, c1 * xv.z);
    atomicAdd(a1 + 3, c1 * xv.w);
}

// ----------------------------- launcher ------------------------------------
extern "C" void kernel_launch(void* const* d_in, const int* in_sizes, int n_in,
                              void* d_out, int out_size, void* d_ws, size_t ws_size,
                              hipStream_t stream) {
    (void)in_sizes; (void)n_in; (void)out_size; (void)ws_size;

    const float* nodes  = (const float*)d_in[0];
    const int*   snd    = (const int*)d_in[1];
    const int*   rcv    = (const int*)d_in[2];
    const float* encW0  = (const float*)d_in[3];
    const float* encb0  = (const float*)d_in[4];
    const float* encW1  = (const float*)d_in[5];
    const float* encb1  = (const float*)d_in[6];
    const float* attW   = (const float*)d_in[7];
    const float* attB   = (const float*)d_in[8];
    const float* coreW  = (const float*)d_in[9];
    const float* coreb  = (const float*)d_in[10];
    const float* decW0  = (const float*)d_in[11];
    const float* decb0  = (const float*)d_in[12];
    const float* decW1  = (const float*)d_in[13];
    const float* decb1  = (const float*)d_in[14];

    // Reference swaps index roles: s, r = receivers, senders.
    const int* sIdx = rcv;
    const int* rIdx = snd;

    char* ws = (char*)d_ws;
    float* x0   = (float*)(ws + 0);             // N*D
    float* x1   = (float*)(ws + 25600000);      // N*D
    float* agg  = (float*)(ws + 51200000);      // N*2D
    float* aS   = (float*)(ws + 102400000);     // N*H
    float* aR   = (float*)(ws + 102800000);     // N*H
    float* smax = (float*)(ws + 103200000);     // N*H
    float* ssum = (float*)(ws + 103600000);     // N*H
    float* wbuf = (float*)(ws + 104000000);     // E*H

    const dim3 gemmBlk(128);
    const dim3 gemmGrd128((GN + 63) / 64, GD / 32);          // Nc = 128 -> 4 col tiles
    const dim3 gemmGrd40((GN + 63) / 64, (GC + 31) / 32);    // Nc = 40  -> 2 col tiles

    // Encoder MLP (relu, relu), K = 128
    gemm_bias_act_wmma<4><<<gemmGrd128, gemmBlk, 0, stream>>>(nodes, encW0, encb0, x1, GN, GD, 1);
    gemm_bias_act_wmma<4><<<gemmGrd128, gemmBlk, 0, stream>>>(x1, encW1, encb1, x0, GN, GD, 1);

    float* xin = x0;
    float* xout = x1;
    for (int h = 0; h < GHOPS; ++h) {
        hop_init<<<(GN * 2 * GD + 255) / 256, 256, 0, stream>>>(smax, ssum, agg, GN * GH, GN * 2 * GD);
        alpha_kernel<<<(GN + 7) / 8, 256, 0, stream>>>(xin, attW + (long)h * GH * 2 * GD, aS, aR, GN);
        edge_logits_max<<<(GE + 255) / 256, 256, 0, stream>>>(aS, aR, sIdx, rIdx,
                                                              attB + (long)h * GH, wbuf, smax, GE);
        edge_exp_sum<<<(GE + 255) / 256, 256, 0, stream>>>(wbuf, rIdx, smax, ssum, GE);
        edge_aggregate<<<(GE + 7) / 8, 256, 0, stream>>>(xin, sIdx, rIdx, wbuf, ssum, agg, GE);
        // Core update: relu(agg @ core_W[h] + core_b[h]), K = 256
        gemm_bias_act_wmma<8><<<gemmGrd128, gemmBlk, 0, stream>>>(agg, coreW + (long)h * (GH * GD) * GD,
                                                                  coreb + (long)h * GD, xout, GN, GD, 1);
        float* t = xin; xin = xout; xout = t;
    }

    // Decoder: Dense+relu, Dense (no activation), K = 128
    gemm_bias_act_wmma<4><<<gemmGrd128, gemmBlk, 0, stream>>>(xin, decW0, decb0, xout, GN, GD, 1);
    gemm_bias_act_wmma<4><<<gemmGrd40, gemmBlk, 0, stream>>>(xout, decW1, decb1, (float*)d_out, GN, GC, 0);
}
